// GAT_35983236006136
// MI455X (gfx1250) — compile-verified, matrix operand order checked by reading, not run on hardware
//
#include <hip/hip_runtime.h>
#include <hip/hip_bf16.h>

// ---------------------------------------------------------------------------
// Types for WMMA
// ---------------------------------------------------------------------------
typedef __attribute__((ext_vector_type(16))) __bf16 bf16x16;
typedef __attribute__((ext_vector_type(8)))  float  f32x8;

union Frag { uint4 q[2]; bf16x16 f; };   // 32 bytes

#define NEG_SLOPE 0.2f
#define EPS 1e-16f

// fp32 -> bf16 bits (round to nearest even)
__device__ __forceinline__ unsigned bf16bits(float f) {
    union { float f; unsigned u; } v; v.f = f;
    return (v.u + 0x7FFFu + ((v.u >> 16) & 1u)) >> 16;
}

// monotone float<->uint mapping for atomicMax on signed floats
__device__ __forceinline__ unsigned f2ord(float f) {
    unsigned u = __float_as_uint(f);
    return (u & 0x80000000u) ? ~u : (u | 0x80000000u);
}
__device__ __forceinline__ float ord2f(unsigned u) {
    return (u & 0x80000000u) ? __uint_as_float(u & 0x7FFFFFFFu)
                             : __uint_as_float(~u);
}
#define ORD_NEG_INF 0x007FFFFFu   // f2ord(-inf)

// ---------------------------------------------------------------------------
// Fill / convert / pack kernels
// ---------------------------------------------------------------------------
__global__ void fill_f32_k(float* p, float v, long long n) {
    long long i = (long long)blockIdx.x * blockDim.x + threadIdx.x;
    if (i < n) p[i] = v;
}
__global__ void fill_u32_k(unsigned* p, unsigned v, long long n) {
    long long i = (long long)blockIdx.x * blockDim.x + threadIdx.x;
    if (i < n) p[i] = v;
}

// fp32 -> bf16 row-major copy, 4 elements/thread (n divisible by 4)
__global__ void cvt_bf16x4_k(const float4* __restrict__ in,
                             uint2* __restrict__ out, long long n4) {
    long long i = (long long)blockIdx.x * blockDim.x + threadIdx.x;
    if (i >= n4) return;
    float4 f = in[i];
    uint2 o;
    o.x = bf16bits(f.x) | (bf16bits(f.y) << 16);
    o.y = bf16bits(f.z) | (bf16bits(f.w) << 16);
    out[i] = o;
}

// Pack weight W[K][N] (fp32, row-major) into bf16 WMMA B-fragment order:
// dword index = ((nt*KSTEPS + ks)*32 + lane)*8 + v
// lane&15 = column-in-tile, khalf=(lane>>4)*8; v<4: k=khalf+2v, v>=4: k=16+khalf+2(v-4)
__global__ void pack_w_bf16_k(const float* __restrict__ W, unsigned* __restrict__ Bp,
                              int K, int N) {
    int i = blockIdx.x * blockDim.x + threadIdx.x;
    int KS = K >> 5;
    int total = (N >> 4) * KS * 32 * 8;
    if (i >= total) return;
    int v    = i & 7;
    int lane = (i >> 3) & 31;
    int ks   = (i >> 8) % KS;
    int nt   = (i >> 8) / KS;
    int col   = nt * 16 + (lane & 15);
    int khalf = (lane >> 4) << 3;
    int k = (ks << 5) + ((v < 4) ? (khalf + 2 * v) : (16 + khalf + 2 * (v - 4)));
    unsigned lo = bf16bits(W[(long long)k * N + col]);
    unsigned hi = bf16bits(W[(long long)(k + 1) * N + col]);
    Bp[i] = lo | (hi << 16);
}

// ---------------------------------------------------------------------------
// WMMA GEMM on pre-converted data: C[M,N] = Abf[M,K] * Bp (packed), f32 acc.
// One wave per (MT m-tiles x 1 n-tile). B fragments for all K held in VGPRs.
// A fragment = two contiguous b128 loads from the bf16 row.
// ---------------------------------------------------------------------------
template<int KSTEPS, int MT>
__global__ void wmma_gemm_pk(const unsigned short* __restrict__ Abf,
                             const unsigned* __restrict__ Bp,
                             float* __restrict__ C, int M, int N) {
    int wave = (int)((blockIdx.x * blockDim.x + threadIdx.x) >> 5);
    int lane = threadIdx.x & 31;
    int ntN  = N >> 4;
    int mg   = wave / ntN;
    int nt   = wave % ntN;
    if (mg * (16 * MT) >= M) return;     // wave-uniform: EXEC stays all-1 for WMMA

    const int K = KSTEPS * 32;
    int rc = lane & 15;                  // row-in-tile (A) / col-in-tile (store)
    int kq = (lane >> 4) << 2;           // khalf/2 dword offset: 0 or 4

    Frag b[KSTEPS];
#pragma unroll
    for (int ks = 0; ks < KSTEPS; ++ks) {
        const uint4* bp = (const uint4*)(Bp + (((long long)nt * KSTEPS + ks) * 32 + lane) * 8);
        b[ks].q[0] = bp[0];
        b[ks].q[1] = bp[1];
    }

    f32x8 acc[MT];
#pragma unroll
    for (int m = 0; m < MT; ++m) acc[m] = (f32x8){};

#pragma unroll
    for (int m = 0; m < MT; ++m) {
        long long row = (long long)mg * (16 * MT) + m * 16 + rc;
        const unsigned* adw = (const unsigned*)(Abf + row * K);
#pragma unroll
        for (int ks = 0; ks < KSTEPS; ++ks) {
            Frag a;
            const uint4* ap = (const uint4*)(adw + ks * 16 + kq);
            a.q[0] = ap[0];
            a.q[1] = ap[2];      // +8 dwords
            acc[m] = __builtin_amdgcn_wmma_f32_16x16x32_bf16(
                false, a.f, false, b[ks].f, (short)0, acc[m], false, false);
        }
    }

    int rbase = (lane >> 4) << 3;
#pragma unroll
    for (int m = 0; m < MT; ++m) {
        float* Cp = C + ((long long)mg * (16 * MT) + m * 16 + rbase) * N + nt * 16 + rc;
#pragma unroll
        for (int v = 0; v < 8; ++v)
            Cp[(long long)v * N] = acc[m][v];
    }
}

// ---------------------------------------------------------------------------
// alpha_src[n,h] = sum_c h[n,h,c]*a_src[h,c] ; alpha_dst likewise
// ---------------------------------------------------------------------------
template<int H, int Cc>
__global__ void alpha_k(const float* __restrict__ h,
                        const float* __restrict__ a_src,
                        const float* __restrict__ a_dst,
                        float* __restrict__ as, float* __restrict__ ad,
                        long long Nn) {
    long long i = (long long)blockIdx.x * blockDim.x + threadIdx.x;
    if (i >= Nn * H) return;
    long long n = i / H;
    int hd = (int)(i % H);
    const float* hp = h + (n * H + hd) * Cc;
    float s = 0.f, d = 0.f;
#pragma unroll
    for (int c = 0; c < Cc; ++c) {
        float hv = hp[c];
        s += hv * a_src[hd * Cc + c];
        d += hv * a_dst[hd * Cc + c];
    }
    as[i] = s;
    ad[i] = d;
}

// ---------------------------------------------------------------------------
// Edge passes; edge ids >= E are self loops (node = e - E)
// ---------------------------------------------------------------------------
template<int H>
__global__ void edge_max_k(const int* __restrict__ src, const int* __restrict__ dst,
                           const float* __restrict__ as, const float* __restrict__ ad,
                           unsigned* __restrict__ emax, long long E, long long Nn) {
    long long i = (long long)blockIdx.x * blockDim.x + threadIdx.x;
    if (i >= (E + Nn) * H) return;
    long long e = i / H;
    int hd = (int)(i % H);
    int s, d;
    if (e < E) { s = src[e]; d = dst[e]; } else { s = d = (int)(e - E); }
    float v = as[(long long)s * H + hd] + ad[(long long)d * H + hd];
    v = v > 0.f ? v : v * NEG_SLOPE;
    atomicMax(emax + (long long)d * H + hd, f2ord(v));
}

template<int H>
__global__ void edge_sum_k(const int* __restrict__ src, const int* __restrict__ dst,
                           const float* __restrict__ as, const float* __restrict__ ad,
                           const unsigned* __restrict__ emax,
                           float* __restrict__ denom, long long E, long long Nn) {
    long long i = (long long)blockIdx.x * blockDim.x + threadIdx.x;
    if (i >= (E + Nn) * H) return;
    long long e = i / H;
    int hd = (int)(i % H);
    int s, d;
    if (e < E) { s = src[e]; d = dst[e]; } else { s = d = (int)(e - E); }
    float v = as[(long long)s * H + hd] + ad[(long long)d * H + hd];
    v = v > 0.f ? v : v * NEG_SLOPE;
    float ex = __expf(v - ord2f(emax[(long long)d * H + hd]));
    atomicAdd(denom + (long long)d * H + hd, ex);
}

template<int H, int Cc>
__global__ void edge_agg_k(const int* __restrict__ src, const int* __restrict__ dst,
                           const float* __restrict__ as, const float* __restrict__ ad,
                           const unsigned* __restrict__ emax,
                           const float* __restrict__ denom,
                           const float* __restrict__ hfeat,
                           float* __restrict__ agg, long long E, long long Nn) {
    const int HC = H * Cc;
    long long i = (long long)blockIdx.x * blockDim.x + threadIdx.x;
    if (i >= (E + Nn) * HC) return;
    long long e = i / HC;
    int idx = (int)(i % HC);
    int hd = idx / Cc;
    int s, d;
    if (e < E) { s = src[e]; d = dst[e]; } else { s = d = (int)(e - E); }
    float v = as[(long long)s * H + hd] + ad[(long long)d * H + hd];
    v = v > 0.f ? v : v * NEG_SLOPE;
    float ex = __expf(v - ord2f(emax[(long long)d * H + hd]));
    float att = ex / (denom[(long long)d * H + hd] + EPS);
    atomicAdd(agg + (long long)d * HC + idx, att * hfeat[(long long)s * HC + idx]);
}

// layer-1 epilogue: in place  agg[i] = elu(agg[i] + b1[i % F])
template<int F>
__global__ void bias_elu_k(float* __restrict__ agg, const float* __restrict__ bias,
                           long long Nn) {
    long long i = (long long)blockIdx.x * blockDim.x + threadIdx.x;
    if (i >= Nn * F) return;
    float v = agg[i] + bias[(int)(i % F)];
    agg[i] = v > 0.f ? v : (__expf(v) - 1.f);
}

// layer-2 epilogue: mean over heads + bias, then log_softmax over NC classes
template<int H, int NC>
__global__ void final_k(const float* __restrict__ agg2, const float* __restrict__ b2,
                        float* __restrict__ out, long long Nn) {
    long long n = (long long)blockIdx.x * blockDim.x + threadIdx.x;
    if (n >= Nn) return;
    float logits[NC];
    const float* p = agg2 + n * (long long)(H * NC);
#pragma unroll
    for (int c = 0; c < NC; ++c) {
        float m = 0.f;
#pragma unroll
        for (int hd = 0; hd < H; ++hd) m += p[hd * NC + c];
        logits[c] = m / (float)H + b2[c];
    }
    float mx = logits[0];
#pragma unroll
    for (int c = 1; c < NC; ++c) mx = fmaxf(mx, logits[c]);
    float ssum = 0.f;
#pragma unroll
    for (int c = 0; c < NC; ++c) ssum += __expf(logits[c] - mx);
    float lse = mx + __logf(ssum);
#pragma unroll
    for (int c = 0; c < NC; ++c) out[n * NC + c] = logits[c] - lse;
}

// ---------------------------------------------------------------------------
// Host launcher
// ---------------------------------------------------------------------------
static inline int gblocks(long long total, int bs) {
    return (int)((total + bs - 1) / bs);
}

extern "C" void kernel_launch(void* const* d_in, const int* in_sizes, int n_in,
                              void* d_out, int out_size, void* d_ws, size_t ws_size,
                              hipStream_t stream) {
    (void)in_sizes; (void)n_in; (void)out_size; (void)ws_size;

    const long long Nn = 100000, E = 1600000;
    const int H = 8, C1 = 8, NC = 20, Fin = 128;
    const int F1 = H * C1;     // 64
    const int F2 = H * NC;     // 160
    const int MT = 5;          // 6250 m-tiles = 1250 groups of 5

    const float* x        = (const float*)d_in[0];
    const int*   src      = (const int*)d_in[1];            // edge_index[0]
    const int*   dst      = src + E;                        // edge_index[1]
    const float* w1       = (const float*)d_in[2];
    const float* att_src1 = (const float*)d_in[3];
    const float* att_dst1 = (const float*)d_in[4];
    const float* b1       = (const float*)d_in[5];
    const float* w2       = (const float*)d_in[6];
    const float* att_src2 = (const float*)d_in[7];
    const float* att_dst2 = (const float*)d_in[8];
    const float* b2       = (const float*)d_in[9];
    float* out = (float*)d_out;

    // workspace layout (in float units; bf16 buffers use half counts)
    float* base = (float*)d_ws;
    long long off = 0;
    float*    h1    = base + off; off += Nn * F1;
    float*    as1   = base + off; off += Nn * H;
    float*    ad1   = base + off; off += Nn * H;
    unsigned* emax1 = (unsigned*)(base + off); off += Nn * H;
    float*    den1  = base + off; off += Nn * H;
    float*    agg1  = base + off; off += Nn * F1;           // becomes h1act in place
    float*    h2    = base + off; off += Nn * F2;
    float*    as2   = base + off; off += Nn * H;
    float*    ad2   = base + off; off += Nn * H;
    unsigned* emax2 = (unsigned*)(base + off); off += Nn * H;
    float*    den2  = base + off; off += Nn * H;
    float*    agg2  = base + off; off += Nn * F2;
    unsigned short* xbf   = (unsigned short*)(base + off); off += Nn * Fin / 2;
    unsigned short* h1abf = (unsigned short*)(base + off); off += Nn * F1 / 2;
    unsigned* w1p = (unsigned*)(base + off); off += (F1 / 16) * (Fin / 32) * 256;
    unsigned* w2p = (unsigned*)(base + off); off += (F2 / 16) * (F1 / 32) * 256;

    const int BS = 256;

    // init (every call: harness does not re-poison between graph replays)
    fill_u32_k<<<gblocks(Nn * H, BS), BS, 0, stream>>>(emax1, ORD_NEG_INF, Nn * H);
    fill_u32_k<<<gblocks(Nn * H, BS), BS, 0, stream>>>(emax2, ORD_NEG_INF, Nn * H);
    fill_f32_k<<<gblocks(Nn * H, BS), BS, 0, stream>>>(den1, 0.f, Nn * H);
    fill_f32_k<<<gblocks(Nn * H, BS), BS, 0, stream>>>(den2, 0.f, Nn * H);
    fill_f32_k<<<gblocks(Nn * F1, BS), BS, 0, stream>>>(agg1, 0.f, Nn * F1);
    fill_f32_k<<<gblocks(Nn * F2, BS), BS, 0, stream>>>(agg2, 0.f, Nn * F2);

    // ---- layer 1 ----
    cvt_bf16x4_k<<<gblocks(Nn * Fin / 4, BS), BS, 0, stream>>>(
        (const float4*)x, (uint2*)xbf, Nn * Fin / 4);
    pack_w_bf16_k<<<gblocks((F1 / 16) * (Fin / 32) * 256, BS), BS, 0, stream>>>(
        w1, w1p, Fin, F1);
    {   // h1 = x @ w1 : M=100000, K=128 (KSTEPS=4), N=64
        long long waves = (Nn / (16 * MT)) * (F1 / 16);
        wmma_gemm_pk<4, MT><<<gblocks(waves * 32, BS), BS, 0, stream>>>(
            xbf, w1p, h1, (int)Nn, F1);
    }
    alpha_k<8, 8><<<gblocks(Nn * H, BS), BS, 0, stream>>>(h1, att_src1, att_dst1,
                                                          as1, ad1, Nn);
    edge_max_k<8><<<gblocks((E + Nn) * H, BS), BS, 0, stream>>>(src, dst, as1, ad1,
                                                                emax1, E, Nn);
    edge_sum_k<8><<<gblocks((E + Nn) * H, BS), BS, 0, stream>>>(src, dst, as1, ad1,
                                                                emax1, den1, E, Nn);
    edge_agg_k<8, 8><<<gblocks((E + Nn) * 64, BS), BS, 0, stream>>>(
        src, dst, as1, ad1, emax1, den1, h1, agg1, E, Nn);
    bias_elu_k<64><<<gblocks(Nn * F1, BS), BS, 0, stream>>>(agg1, b1, Nn);

    // ---- layer 2 ----
    cvt_bf16x4_k<<<gblocks(Nn * F1 / 4, BS), BS, 0, stream>>>(
        (const float4*)agg1, (uint2*)h1abf, Nn * F1 / 4);
    pack_w_bf16_k<<<gblocks((F2 / 16) * (F1 / 32) * 256, BS), BS, 0, stream>>>(
        w2, w2p, F1, F2);
    {   // h2 = h1act @ w2 : M=100000, K=64 (KSTEPS=2), N=160
        long long waves = (Nn / (16 * MT)) * (F2 / 16);
        wmma_gemm_pk<2, MT><<<gblocks(waves * 32, BS), BS, 0, stream>>>(
            h1abf, w2p, h2, (int)Nn, F2);
    }
    alpha_k<8, 20><<<gblocks(Nn * H, BS), BS, 0, stream>>>(h2, att_src2, att_dst2,
                                                           as2, ad2, Nn);
    edge_max_k<8><<<gblocks((E + Nn) * H, BS), BS, 0, stream>>>(src, dst, as2, ad2,
                                                                emax2, E, Nn);
    edge_sum_k<8><<<gblocks((E + Nn) * H, BS), BS, 0, stream>>>(src, dst, as2, ad2,
                                                                emax2, den2, E, Nn);
    edge_agg_k<8, 20><<<gblocks((E + Nn) * 160, BS), BS, 0, stream>>>(
        src, dst, as2, ad2, emax2, den2, h2, agg2, E, Nn);
    final_k<8, 20><<<gblocks(Nn, BS), BS, 0, stream>>>(agg2, b2, out, Nn);
}
